// ViMHA_2937757630685
// MI455X (gfx1250) — matrix operand order repaired and checked by reading, not compiled, not run on hardware
//
#include <hip/hip_runtime.h>

// ---------------------------------------------------------------------------
// Types for CDNA5 WMMA (wave32): D(16x16 f32) = A(16x32 f16) x B(32x16 f16) + C
// ---------------------------------------------------------------------------
typedef _Float16 h16;
typedef __attribute__((ext_vector_type(8)))  _Float16 v8h;
typedef __attribute__((ext_vector_type(16))) _Float16 v16h;
typedef __attribute__((ext_vector_type(8)))  float    v8f;
typedef __attribute__((ext_vector_type(4)))  int      v4i;

#define VWMMA(a, b, c)                                                         \
  __builtin_amdgcn_wmma_f32_16x16x32_f16(false, (a), false, (b), (short)0,     \
                                         (c), false, false)

static __device__ __forceinline__ v16h cat8(v8h lo, v8h hi) {
  return __builtin_shufflevector(lo, hi, 0, 1, 2, 3, 4, 5, 6, 7, 8, 9, 10, 11,
                                 12, 13, 14, 15);
}

// --------------------- gfx1250 async-copy path (verified) ------------------
#if defined(__has_builtin)
#if __has_builtin(__builtin_amdgcn_global_load_async_to_lds_b128)
#define HAVE_ASYNC_LDS 1
#endif
#endif

#ifdef HAVE_ASYNC_LDS
// diag-verified param types: (int4 AS1*, int4 AS3*, imm, imm)
typedef __attribute__((address_space(1))) v4i as1_v4i;
typedef __attribute__((address_space(3))) v4i as3_v4i;
static __device__ __forceinline__ void async_b128(const h16* g, h16* l) {
  __builtin_amdgcn_global_load_async_to_lds_b128((as1_v4i*)g, (as3_v4i*)l, 0, 0);
}
static __device__ __forceinline__ void wait_async0() {
#if __has_builtin(__builtin_amdgcn_s_wait_asynccnt)
  __builtin_amdgcn_s_wait_asynccnt(0);
#else
  asm volatile("s_wait_asynccnt 0" ::: "memory");
#endif
}
static __device__ __forceinline__ void wait_async6() {
#if __has_builtin(__builtin_amdgcn_s_wait_asynccnt)
  __builtin_amdgcn_s_wait_asynccnt(6);
#else
  asm volatile("s_wait_asynccnt 6" ::: "memory");
#endif
}
#endif

// ---------------------------------------------------------------------------
// Problem constants
// ---------------------------------------------------------------------------
// x: (8,4,512,512) f32; W: (3072,1024) f32; P=16, heads=16, n=1024, d=1024
#define NB     8
#define NTOK   1024
#define DMODEL 1024
#define NHEADS 16
#define DHEAD  64
#define QK_SCALE 0.03125f            // 1/sqrt(1024)
#define LN10000 9.210340371976184f

// ---------------------------------------------------------------------------
// Kernel 1: W (3072x1024 f32, torch Linear layout) -> f16, same layout
// ---------------------------------------------------------------------------
__global__ void vimha_wcvt(const float* __restrict__ W, h16* __restrict__ Wh) {
  int i = blockIdx.x * blockDim.x + threadIdx.x;   // 3 * 1024 * 1024 elements
  Wh[i] = (h16)W[i];
}

// ---------------------------------------------------------------------------
// Kernel 2: build tokens:  tok[b,i,j] = 2*x[patch-permuted] + pe[i,j]  (f16)
// ---------------------------------------------------------------------------
__global__ void vimha_tokens(const float* __restrict__ x, h16* __restrict__ tok) {
  int idx = blockIdx.x * blockDim.x + threadIdx.x;        // 8 * 1024 * 1024
  int b = idx >> 20;
  int f = idx & 0xFFFFF;
  int i = f >> 10;
  int j = f & 1023;
  int c      = f >> 18;
  int rem    = f & 0x3FFFF;
  int patch  = rem >> 8;
  int within = rem & 255;
  int r = within >> 4, cc = within & 15;
  int prow = patch >> 5, pcol = patch & 31;
  int src = ((b * 4 + c) * 512 + prow * 16 + r) * 512 + pcol * 16 + cc;
  float jj   = (float)(j & ~1);
  float freq = __expf(jj * (-LN10000 / 1024.0f));
  float ang  = (float)i * freq;
  float pe   = (j & 1) ? __cosf(ang) : __sinf(ang);
  tok[idx] = (h16)(2.0f * x[src] + pe);
}

// ---------------------------------------------------------------------------
// Kernel 3: per-batch GEMM qkv = tok @ W^T   (M=1024, N=3072, K=1024)
// 128 thr (4 waves), tile 128x64, K-step 32, DOUBLE-BUFFERED async-to-LDS:
// issue tile t+1 (6 async instrs/wave), s_wait_asynccnt 6 retires tile t
// (in-order completion), barrier, compute 8 WMMA while t+1 streams in.
// ---------------------------------------------------------------------------
#define BM   128
#define BN   64
#define BK   32
#define LDSK 40   // padded half-stride (80B, 16B aligned, conflict-free)
#define KTILES (DMODEL / BK)

// Uniform, divergence-free tile loader: 4 A-vectors + 2 B-vectors per thread.
static __device__ __forceinline__ void load_tiles(
    const h16* __restrict__ Abase, const h16* __restrict__ Wh,
    h16* As, h16* Bs, int m0, int n0, int k0, int tid) {
#pragma unroll
  for (int v = 0; v < 4; ++v) {
    int vec = tid + v * 128;                    // 0..511
    int row = vec >> 2, c8 = (vec & 3) * 8;
    const h16* g = &Abase[(size_t)(m0 + row) * DMODEL + k0 + c8];
    h16*       l = &As[row * LDSK + c8];
#ifdef HAVE_ASYNC_LDS
    async_b128(g, l);
#else
    *(v8h*)l = *(const v8h*)g;
#endif
  }
#pragma unroll
  for (int v = 0; v < 2; ++v) {
    int vec = tid + v * 128;                    // 0..255
    int row = vec >> 2, c8 = (vec & 3) * 8;
    const h16* g = &Wh[(size_t)(n0 + row) * DMODEL + k0 + c8];
    h16*       l = &Bs[row * LDSK + c8];
#ifdef HAVE_ASYNC_LDS
    async_b128(g, l);
#else
    *(v8h*)l = *(const v8h*)g;
#endif
  }
}

__global__ __launch_bounds__(128) void vimha_qkv(
    const h16* __restrict__ tok,    // (8,1024,1024)
    const h16* __restrict__ Wh,     // (3072,1024)
    h16* __restrict__ qbuf,         // (8,16,1024,64)
    h16* __restrict__ kbuf,         // (8,16,1024,64)
    h16* __restrict__ vtbuf)        // (8,16,64,1024)
{
  __shared__ __align__(16) h16 As[2][BM * LDSK];   // 2 x 10240 B
  __shared__ __align__(16) h16 Bs[2][BN * LDSK];   // 2 x  5120 B

  const int b    = blockIdx.z;
  const int m0   = blockIdx.y * BM;
  const int n0   = blockIdx.x * BN;
  const int tid  = threadIdx.x;
  const int lane = tid & 31;
  const int wave = tid >> 5;
  const int m15  = lane & 15;
  const int hs   = lane >> 4;

  const h16* Abase = tok + (size_t)b * NTOK * DMODEL;

  v8f acc[2][4] = {};

  // prologue: tile 0 -> buffer 0
  load_tiles(Abase, Wh, As[0], Bs[0], m0, n0, 0, tid);

#pragma unroll 2
  for (int t = 0; t < KTILES; ++t) {
    const int cb = t & 1;
    if (t + 1 < KTILES) {
      load_tiles(Abase, Wh, As[cb ^ 1], Bs[cb ^ 1], m0, n0, (t + 1) * BK, tid);
#ifdef HAVE_ASYNC_LDS
      wait_async6();     // retire tile t's 6 loads; tile t+1 stays in flight
#endif
    } else {
#ifdef HAVE_ASYNC_LDS
      wait_async0();
#endif
    }
    __syncthreads();     // all waves' tile-t writes visible

    // Preload ALL fragments, then fire the WMMA burst (staggered dscnt waits).
    // A fragment (16x32): lanes 0-15 K=0-15, lanes 16-31 K=16-31
    const int kb = hs ? 16 : 0;
    v16h af[2];
#pragma unroll
    for (int s = 0; s < 2; ++s) {
      int mrow = s * 64 + wave * 16 + m15;
      af[s] = cat8(*(const v8h*)&As[cb][mrow * LDSK + kb],
                   *(const v8h*)&As[cb][mrow * LDSK + kb + 8]);
    }
    // B fragments (32x16): lanes 0-15 K=0-15, lanes 16-31 K=16-31
    const int klo = hs ? 16 : 0;
    v16h bf[4];
#pragma unroll
    for (int ct = 0; ct < 4; ++ct) {
      int col = ct * 16 + m15;
      bf[ct] = cat8(*(const v8h*)&Bs[cb][col * LDSK + klo],
                    *(const v8h*)&Bs[cb][col * LDSK + klo + 8]);
    }
#pragma unroll
    for (int s = 0; s < 2; ++s)
#pragma unroll
      for (int ct = 0; ct < 4; ++ct)
        acc[s][ct] = VWMMA(af[s], bf[ct], acc[s][ct]);
    __syncthreads();     // safe to overwrite this buffer next round
  }

  // epilogue: C layout row = vr + 8*(lane>=16), col = lane&15
#pragma unroll
  for (int s = 0; s < 2; ++s) {
#pragma unroll
    for (int ct = 0; ct < 4; ++ct) {
      int o = n0 + ct * 16 + m15;
#pragma unroll
      for (int vr = 0; vr < 8; ++vr) {
        int   i   = m0 + s * 64 + wave * 16 + vr + hs * 8;
        float val = acc[s][ct][vr];
        if (o < 1024) {                     // Q (pre-scaled)
          int h = o >> 6, dhi = o & 63;
          qbuf[(((size_t)b * NHEADS + h) * NTOK + i) * DHEAD + dhi] =
              (h16)(val * QK_SCALE);
        } else if (o < 2048) {              // K
          int oo = o - 1024, h = oo >> 6, dhi = oo & 63;
          kbuf[(((size_t)b * NHEADS + h) * NTOK + i) * DHEAD + dhi] = (h16)val;
        } else {                            // V transposed (dh-major)
          int oo = o - 2048, h = oo >> 6, dhi = oo & 63;
          vtbuf[(((size_t)b * NHEADS + h) * DHEAD + dhi) * NTOK + i] = (h16)val;
        }
      }
    }
  }
}

// ---------------------------------------------------------------------------
// Kernel 4: flash attention, one wave per (b, head, 16-query block).
// S = Q K^T (WMMA), online softmax (16-lane butterfly reductions),
// P re-layout via LDS, O += P V (WMMA), scatter to image layout (f32 out).
// ---------------------------------------------------------------------------
#define PLDS 40   // padded half-stride for P tile

__global__ __launch_bounds__(32) void vimha_attn(
    const h16* __restrict__ qbuf,
    const h16* __restrict__ kbuf,
    const h16* __restrict__ vtbuf,
    float* __restrict__ out)
{
  __shared__ __align__(16) h16 pls[16 * PLDS];

  const int lane = threadIdx.x;
  const int q0   = blockIdx.x * 16;
  const int h    = blockIdx.y;
  const int b    = blockIdx.z;
  const int m15  = lane & 15;
  const int hs   = lane >> 4;

  const h16* qp = qbuf + ((size_t)b * NHEADS + h) * NTOK * DHEAD;
  const h16* kp = kbuf + ((size_t)b * NHEADS + h) * NTOK * DHEAD;
  const h16* vp = vtbuf + ((size_t)b * NHEADS + h) * DHEAD * NTOK;

  // Q fragments for dh 0-31 and 32-63 (A layout)
  const int  kbq  = hs ? 16 : 0;
  const h16* qrow = qp + (size_t)(q0 + m15) * DHEAD;
  v16h a0 = cat8(*(const v8h*)&qrow[kbq], *(const v8h*)&qrow[kbq + 8]);
  v16h a1 = cat8(*(const v8h*)&qrow[32 + kbq], *(const v8h*)&qrow[32 + kbq + 8]);

  float mrow[8], lrow[8];
  v8f   oacc[4] = {};
#pragma unroll
  for (int r = 0; r < 8; ++r) { mrow[r] = -1e30f; lrow[r] = 0.0f; }

  const int klo = hs ? 16 : 0;           // B-layout K selector per lane-half
  const int kbp = hs ? 16 : 0;           // A-layout K selector for P frag

  for (int j0 = 0; j0 < NTOK; j0 += 32) {
    // ---- S tiles: keys [j0,j0+16) and [j0+16,j0+32), K over dh=64 ----
    v8f s0 = {}, s1 = {};
    {
      const h16* kr0 = kp + (size_t)(j0 + m15) * DHEAD;
      const h16* kr1 = kp + (size_t)(j0 + 16 + m15) * DHEAD;
      v16h bk00 = cat8(*(const v8h*)&kr0[klo], *(const v8h*)&kr0[klo + 8]);
      v16h bk01 = cat8(*(const v8h*)&kr0[32 + klo], *(const v8h*)&kr0[32 + klo + 8]);
      v16h bk10 = cat8(*(const v8h*)&kr1[klo], *(const v8h*)&kr1[klo + 8]);
      v16h bk11 = cat8(*(const v8h*)&kr1[32 + klo], *(const v8h*)&kr1[32 + klo + 8]);
      s0 = VWMMA(a0, bk00, s0);
      s0 = VWMMA(a1, bk01, s0);
      s1 = VWMMA(a0, bk10, s1);
      s1 = VWMMA(a1, bk11, s1);
    }

    // ---- online softmax over the 32 new columns ----
#pragma unroll
    for (int r = 0; r < 8; ++r) {
      float mx = fmaxf(s0[r], s1[r]);
#pragma unroll
      for (int off = 8; off; off >>= 1) mx = fmaxf(mx, __shfl_xor(mx, off, 16));
      float mnew  = fmaxf(mrow[r], mx);
      float alpha = __expf(mrow[r] - mnew);
      float p0    = __expf(s0[r] - mnew);
      float p1    = __expf(s1[r] - mnew);
      float sum   = p0 + p1;
#pragma unroll
      for (int off = 8; off; off >>= 1) sum += __shfl_xor(sum, off, 16);
      lrow[r] = lrow[r] * alpha + sum;
      mrow[r] = mnew;
#pragma unroll
      for (int ct = 0; ct < 4; ++ct) oacc[ct][r] *= alpha;
      int prw = r + hs * 8;              // C-layout row -> LDS row
      pls[prw * PLDS + m15]      = (h16)p0;
      pls[prw * PLDS + 16 + m15] = (h16)p1;
    }
    __syncthreads();                     // S_NOP (1 wave) + compiler DS fence

    // P tile (16x32) back as A-layout fragment
    v16h pfrag = cat8(*(const v8h*)&pls[m15 * PLDS + kbp],
                      *(const v8h*)&pls[m15 * PLDS + kbp + 8]);
    __syncthreads();

    // ---- O += P V : V^T rows are contiguous per output-dh lane ----
#pragma unroll
    for (int ct = 0; ct < 4; ++ct) {
      const h16* vr = vp + (size_t)(ct * 16 + m15) * NTOK + j0 + klo;
      v16h vfrag = cat8(*(const v8h*)&vr[0], *(const v8h*)&vr[8]);
      oacc[ct] = VWMMA(pfrag, vfrag, oacc[ct]);
    }
  }

  // ---- epilogue: normalize and scatter into tokens_to_image layout ----
  float inv[8];
#pragma unroll
  for (int r = 0; r < 8; ++r) inv[r] = 1.0f / lrow[r];
#pragma unroll
  for (int ct = 0; ct < 4; ++ct) {
    int jd = h * DHEAD + ct * 16 + m15;  // column in (B,n,d) out
#pragma unroll
    for (int r = 0; r < 8; ++r) {
      int i    = q0 + r + hs * 8;
      int flat = i * DMODEL + jd;
      int c      = flat >> 18;
      int rem    = flat & 0x3FFFF;
      int patch  = rem >> 8;
      int within = rem & 255;
      int rr = within >> 4, cc = within & 15;
      int prw = patch >> 5, pcl = patch & 31;
      size_t dst = (((size_t)b * 4 + c) * 512 + prw * 16 + rr) * 512 +
                   pcl * 16 + cc;
      out[dst] = oacc[ct][r] * inv[r];
    }
  }
}

// ---------------------------------------------------------------------------
// Launcher
// ---------------------------------------------------------------------------
extern "C" void kernel_launch(void* const* d_in, const int* in_sizes, int n_in,
                              void* d_out, int out_size, void* d_ws,
                              size_t ws_size, hipStream_t stream) {
  (void)in_sizes; (void)n_in; (void)out_size; (void)ws_size;
  const float* x = (const float*)d_in[0];
  const float* W = (const float*)d_in[1];
  float* out = (float*)d_out;

  char* ws = (char*)d_ws;
  // workspace layout (bytes):
  //   Wh   f16 (3072x1024)          @ 0          size  6,291,456
  //   tok  f16 (8x1024x1024)        @ 6,291,456  size 16,777,216
  //   Q    f16 (8x16x1024x64)       @ 23,068,672 size 16,777,216
  //   K    f16 (8x16x1024x64)       @ 39,845,888 size 16,777,216
  //   V^T  f16 (8x16x64x1024)       @ 56,623,104 size 16,777,216  (end 73.4MB)
  h16* Wh  = (h16*)(ws);
  h16* tok = (h16*)(ws + 6291456ull);
  h16* qb  = (h16*)(ws + 23068672ull);
  h16* kb  = (h16*)(ws + 39845888ull);
  h16* vtb = (h16*)(ws + 56623104ull);

  vimha_wcvt<<<dim3(3 * 1024 * 1024 / 256), 256, 0, stream>>>(W, Wh);
  vimha_tokens<<<dim3(8 * 1024 * 1024 / 256), 256, 0, stream>>>(x, tok);
  vimha_qkv<<<dim3(3072 / BN, 1024 / BM, NB), 128, 0, stream>>>(tok, Wh, qb, kb, vtb);
  vimha_attn<<<dim3(NTOK / 16, NHEADS, NB), 32, 0, stream>>>(qb, kb, vtb, out);
}
